// MultiHeadAttentionRoPE_8821862825984
// MI455X (gfx1250) — matrix operand order repaired and checked by reading, not compiled, not run on hardware
//
#include <hip/hip_runtime.h>
#include <hip/hip_bf16.h>
#include <math.h>

#define D_MODEL 1024
#define NH 16
#define HD 64
#define BATCH 2
#define SEQ 2048
#define MTOT (BATCH*SEQ)   // 4096

typedef __attribute__((ext_vector_type(16))) __bf16 v16bf;
typedef __attribute__((ext_vector_type(8)))  float  v8f;

__device__ __forceinline__ int lane_id() { return threadIdx.x & 31; }

// A-matrix 16x32 bf16 frag element -> k index (ISA 7.12.2):
// lanes 0-15 hold K {0..7,16..23}, lanes 16-31 hold K {8..15,24..31}
__device__ __forceinline__ int a_kidx(int e, int kh) {
  return (e & 7) + ((e >> 3) << 4) + (kh << 3);
}

__device__ __forceinline__ v8f wmma_bf16(v16bf a, v16bf b, v8f c) {
  return __builtin_amdgcn_wmma_f32_16x16x32_bf16(false, a, false, b, (short)0, c,
                                                 false, false);
}

// Load A-frag (M=16 rows x K=32) from LDS tile at [row0][k0], row stride in elems.
__device__ __forceinline__ v16bf load_afrag(const __bf16* base, int stride) {
  int l = lane_id(); int r = l & 15; int kh = l >> 4;
  v16bf a;
#pragma unroll
  for (int e = 0; e < 16; ++e) a[e] = base[r * stride + a_kidx(e, kh)];
  return a;
}

// Load B-frag (K=32 x N=16) from LDS holding B^T as [n][k]:
// lane = column n; lanes 0-15 K {0..7,16..23}, lanes 16-31 K {8..15,24..31}.
__device__ __forceinline__ v16bf load_bfrag(const __bf16* base, int stride) {
  int l = lane_id(); int n = l & 15; int kh = l >> 4;
  v16bf b;
#pragma unroll
  for (int e = 0; e < 16; ++e) b[e] = base[n * stride + a_kidx(e, kh)];
  return b;
}

__device__ __forceinline__ float halfmax(float v) {
#pragma unroll
  for (int m = 1; m <= 8; m <<= 1) v = fmaxf(v, __shfl_xor(v, m, 32));
  return v;
}
__device__ __forceinline__ float halfsum(float v) {
#pragma unroll
  for (int m = 1; m <= 8; m <<= 1) v += __shfl_xor(v, m, 32);
  return v;
}

// gfx1250 async copy: each lane moves 16B global -> LDS, tracked by ASYNCcnt.
// VDST = per-lane LDS byte address, VADDR = per-lane 32-bit byte offset,
// SADDR = uniform 64-bit base (GVS mode).
__device__ __forceinline__ void async_g2l_b128(unsigned lds_addr,
                                               const __bf16* gbase,
                                               unsigned goff) {
  asm volatile("global_load_async_to_lds_b128 %0, %1, %2"
               :: "v"(lds_addr), "v"(goff), "s"(gbase)
               : "memory");
}
__device__ __forceinline__ void wait_async0() {
  asm volatile("s_wait_asynccnt 0x0" ::: "memory");
}

// ---------------------------------------------------------------------------
// Kernel 1: QKV projection + bias + RoPE (+ 1/sqrt(d) folded into Q).
// Tile 128(M) x 64(N); N tile == one head. grid=(32,16,3), block=256.
// Q,K stored [B][H][L][d]; V stored TRANSPOSED per head: [B][H][d][L].
// ---------------------------------------------------------------------------
__global__ void __launch_bounds__(256)
qkv_rope_kernel(const float* __restrict__ x,
                const float* __restrict__ Wq, const float* __restrict__ bq,
                const float* __restrict__ Wk, const float* __restrict__ bk,
                const float* __restrict__ Wv, const float* __restrict__ bv,
                __bf16* __restrict__ qout, __bf16* __restrict__ kout,
                __bf16* __restrict__ vout) {
  __shared__ __bf16 aT[128][40];
  __shared__ __bf16 wT[64][40];
  __shared__ float  cT[128][72];

  const int tid  = threadIdx.x;
  const int wid  = tid >> 5;
  const int m0   = blockIdx.x * 128;
  const int head = blockIdx.y;
  const int j0   = head * HD;
  const int sel  = blockIdx.z;

  const float* W    = (sel == 0) ? Wq : (sel == 1) ? Wk : Wv;
  const float* bias = (sel == 0) ? bq : (sel == 1) ? bk : bv;
  __bf16*      dst  = (sel == 0) ? qout : (sel == 1) ? kout : vout;

  const int wm = (wid >> 1) * 32;
  const int wn = (wid & 1) * 32;

  v8f acc[2][2];
#pragma unroll
  for (int i = 0; i < 2; ++i)
#pragma unroll
    for (int j = 0; j < 2; ++j) { v8f z = {}; acc[i][j] = z; }

  for (int k0 = 0; k0 < D_MODEL; k0 += 32) {
    __syncthreads();
    for (int i = tid; i < 128 * 32; i += 256) {
      int r = i >> 5, c = i & 31;
      aT[r][c] = (__bf16)x[(size_t)(m0 + r) * D_MODEL + k0 + c];
    }
    for (int i = tid; i < 64 * 32; i += 256) {
      int r = i >> 5, c = i & 31;
      wT[r][c] = (__bf16)W[(size_t)(j0 + r) * D_MODEL + k0 + c];
    }
    __syncthreads();

    v16bf a0 = load_afrag(&aT[wm][0], 40);
    v16bf a1 = load_afrag(&aT[wm + 16][0], 40);
    v16bf b0 = load_bfrag(&wT[wn][0], 40);
    v16bf b1 = load_bfrag(&wT[wn + 16][0], 40);
    acc[0][0] = wmma_bf16(a0, b0, acc[0][0]);
    acc[0][1] = wmma_bf16(a0, b1, acc[0][1]);
    acc[1][0] = wmma_bf16(a1, b0, acc[1][0]);
    acc[1][1] = wmma_bf16(a1, b1, acc[1][1]);
  }

  __syncthreads();
  {
    int l = lane_id(); int col = l & 15; int rh = (l >> 4) * 8;
#pragma unroll
    for (int ms = 0; ms < 2; ++ms)
#pragma unroll
      for (int ns = 0; ns < 2; ++ns)
#pragma unroll
        for (int r = 0; r < 8; ++r)
          cT[wm + ms * 16 + rh + r][wn + ns * 16 + col] = acc[ms][ns][r];
  }
  __syncthreads();

  for (int i = tid; i < 128 * 64; i += 256) {
    int r = i >> 6, t = i & 63;
    float v = cT[r][t] + bias[j0 + t];
    int m = m0 + r;
    int b = m >> 11;          // / SEQ
    int l = m & (SEQ - 1);
    if (sel < 2) {
      float p  = cT[r][t ^ 32] + bias[j0 + (t ^ 32)];
      int   fi = t & 31;
      float inv = __powf(10000.0f, -(2.0f * (float)fi) / 64.0f);
      float ang = (float)l * inv;
      float sn, cs;
      __sincosf(ang, &sn, &cs);
      float outv = (t < 32) ? (v * cs - p * sn) : (v * cs + p * sn);
      if (sel == 0) outv *= 0.125f;   // 1/sqrt(64) folded into Q
      dst[(((size_t)b * NH + head) * SEQ + l) * HD + t] = (__bf16)outv;
    } else {
      // V transposed: [B][H][d][L]
      dst[(((size_t)b * NH + head) * HD + t) * SEQ + l] = (__bf16)v;
    }
  }
}

// ---------------------------------------------------------------------------
// Kernel 2: causal flash attention. grid=(SEQ/128, NH, BATCH), block=256.
// Each wave owns 16 query rows; key blocks of 64; online softmax.
// Q/K/V^T tiles filled with GLOBAL_LOAD_ASYNC_TO_LDS_B128 (ASYNCcnt).
// ---------------------------------------------------------------------------
__global__ void __launch_bounds__(256)
attn_kernel(const __bf16* __restrict__ qb, const __bf16* __restrict__ kb,
            const __bf16* __restrict__ vb, __bf16* __restrict__ ctx) {
  __shared__ __bf16 qT[128][72];
  __shared__ __bf16 kT[64][72];
  __shared__ __bf16 vtT[64][72];     // V^T tile: [dcol][key]
  __shared__ __bf16 pT[8][16][72];   // per-wave P staging (C-layout -> A-frag)

  const int tid   = threadIdx.x;
  const int wid   = tid >> 5;
  const int lane  = tid & 31;
  const int col   = lane & 15;
  const int rh    = (lane >> 4) * 8;
  const int qbase = blockIdx.x * 128;
  const int h     = blockIdx.y;
  const int b     = blockIdx.z;

  const __bf16* Q  = qb + (size_t)(b * NH + h) * SEQ * HD;  // [L][d]
  const __bf16* K  = kb + (size_t)(b * NH + h) * SEQ * HD;  // [L][d]
  const __bf16* Vt = vb + (size_t)(b * NH + h) * HD * SEQ;  // [d][L]

  // async-fill Q tile: 128 rows x 64 cols = 128 rows x 8 b128 chunks
  for (int i = tid; i < 128 * 8; i += 256) {
    int r = i >> 3, c = (i & 7) * 8;
    async_g2l_b128((unsigned)(uintptr_t)&qT[r][c], Q,
                   (unsigned)(((qbase + r) * HD + c) * sizeof(__bf16)));
  }
  wait_async0();
  __syncthreads();

  v16bf qf[2];
  qf[0] = load_afrag(&qT[wid * 16][0], 72);
  qf[1] = load_afrag(&qT[wid * 16][32], 72);

  v8f o[4];
#pragma unroll
  for (int ns = 0; ns < 4; ++ns) { v8f z = {}; o[ns] = z; }
  float rowm[8], rowl[8];
#pragma unroll
  for (int r = 0; r < 8; ++r) { rowm[r] = -3.0e38f; rowl[r] = 0.0f; }

  const int nkb = (qbase >> 6) + 2;   // causal: key blocks 0..nkb-1
  for (int jb = 0; jb < nkb; ++jb) {
    const int kb0 = jb * 64;
    __syncthreads();    // previous iteration's K/V^T reads are done
    // K tile: 64 rows x 8 chunks; V^T tile: 64 d-rows x 8 chunks (64 keys)
    for (int i = tid; i < 64 * 8; i += 256) {
      int r = i >> 3, c = (i & 7) * 8;
      async_g2l_b128((unsigned)(uintptr_t)&kT[r][c], K,
                     (unsigned)(((kb0 + r) * HD + c) * sizeof(__bf16)));
      async_g2l_b128((unsigned)(uintptr_t)&vtT[r][c], Vt,
                     (unsigned)((r * SEQ + kb0 + c) * sizeof(__bf16)));
    }
    if (jb + 1 < nkb) {   // nudge next K block toward L2 (global_prefetch_b8)
      __builtin_prefetch(&K[(size_t)(kb0 + 64 + (tid & 63)) * HD], 0, 0);
    }
    wait_async0();
    __syncthreads();

    // S = Q K^T (scale already folded into Q)
    v8f s[4];
#pragma unroll
    for (int ns = 0; ns < 4; ++ns) {
      v8f c = {};
      v16bf bk0 = load_bfrag(&kT[ns * 16][0], 72);
      v16bf bk1 = load_bfrag(&kT[ns * 16][32], 72);
      c = wmma_bf16(qf[0], bk0, c);
      c = wmma_bf16(qf[1], bk1, c);
      s[ns] = c;
    }

    // causal mask + running row max
    float nm[8];
#pragma unroll
    for (int r = 0; r < 8; ++r) {
      int qrow = qbase + wid * 16 + rh + r;
      float lm = -3.0e38f;
#pragma unroll
      for (int ns = 0; ns < 4; ++ns) {
        int key = kb0 + ns * 16 + col;
        float sv = s[ns][r];
        if (key > qrow) { sv = -3.0e38f; s[ns][r] = sv; }
        lm = fmaxf(lm, sv);
      }
      lm = halfmax(lm);
      nm[r] = fmaxf(rowm[r], lm);
    }

    // P = exp(S - m_new); rescale O and l
#pragma unroll
    for (int r = 0; r < 8; ++r) {
      float corr = __expf(rowm[r] - nm[r]);
      rowm[r] = nm[r];
      float ps = 0.0f;
#pragma unroll
      for (int ns = 0; ns < 4; ++ns) {
        float p = __expf(s[ns][r] - nm[r]);
        ps += p;
        pT[wid][rh + r][ns * 16 + col] = (__bf16)p;
      }
#pragma unroll
      for (int ns = 0; ns < 4; ++ns) o[ns][r] *= corr;
      rowl[r] = rowl[r] * corr + halfsum(ps);
    }

    // O += P @ V   (per-wave LDS round-trip; DS ops are in-order per wave)
#pragma unroll
    for (int kc = 0; kc < 2; ++kc) {
      v16bf pa = load_afrag(&pT[wid][0][kc * 32], 72);
#pragma unroll
      for (int ns = 0; ns < 4; ++ns) {
        v16bf vf = load_bfrag(&vtT[ns * 16][kc * 32], 72);
        o[ns] = wmma_bf16(pa, vf, o[ns]);
      }
    }
  }

  // normalize and store context: [B][L][H*HD] bf16
#pragma unroll
  for (int ns = 0; ns < 4; ++ns)
#pragma unroll
    for (int r = 0; r < 8; ++r) {
      int qrow = qbase + wid * 16 + rh + r;
      float val = o[ns][r] / rowl[r];
      ctx[((size_t)b * SEQ + qrow) * D_MODEL + h * HD + ns * 16 + col] =
          (__bf16)val;
    }
}

// ---------------------------------------------------------------------------
// Kernel 3: out = ctx @ Wo^T + bo (fp32 out). grid=(32,16), block=256.
// ---------------------------------------------------------------------------
__global__ void __launch_bounds__(256)
oproj_kernel(const __bf16* __restrict__ ctx, const float* __restrict__ Wo,
             const float* __restrict__ bo, float* __restrict__ out) {
  __shared__ __bf16 aT[128][40];
  __shared__ __bf16 wT[64][40];

  const int tid = threadIdx.x;
  const int wid = tid >> 5;
  const int m0  = blockIdx.x * 128;
  const int j0  = blockIdx.y * 64;
  const int wm  = (wid >> 1) * 32;
  const int wn  = (wid & 1) * 32;

  v8f acc[2][2];
#pragma unroll
  for (int i = 0; i < 2; ++i)
#pragma unroll
    for (int j = 0; j < 2; ++j) { v8f z = {}; acc[i][j] = z; }

  for (int k0 = 0; k0 < D_MODEL; k0 += 32) {
    __syncthreads();
    for (int i = tid; i < 128 * 32; i += 256) {
      int r = i >> 5, c = i & 31;
      aT[r][c] = ctx[(size_t)(m0 + r) * D_MODEL + k0 + c];
    }
    for (int i = tid; i < 64 * 32; i += 256) {
      int r = i >> 5, c = i & 31;
      wT[r][c] = (__bf16)Wo[(size_t)(j0 + r) * D_MODEL + k0 + c];
    }
    __syncthreads();

    v16bf a0 = load_afrag(&aT[wm][0], 40);
    v16bf a1 = load_afrag(&aT[wm + 16][0], 40);
    v16bf b0 = load_bfrag(&wT[wn][0], 40);
    v16bf b1 = load_bfrag(&wT[wn + 16][0], 40);
    acc[0][0] = wmma_bf16(a0, b0, acc[0][0]);
    acc[0][1] = wmma_bf16(a0, b1, acc[0][1]);
    acc[1][0] = wmma_bf16(a1, b0, acc[1][0]);
    acc[1][1] = wmma_bf16(a1, b1, acc[1][1]);
  }

  int l = lane_id(); int col = l & 15; int rh = (l >> 4) * 8;
#pragma unroll
  for (int ms = 0; ms < 2; ++ms)
#pragma unroll
    for (int ns = 0; ns < 2; ++ns)
#pragma unroll
      for (int r = 0; r < 8; ++r) {
        int row  = m0 + wm + ms * 16 + rh + r;
        int jcol = j0 + wn + ns * 16 + col;
        out[(size_t)row * D_MODEL + jcol] = acc[ms][ns][r] + bo[jcol];
      }
}

// ---------------------------------------------------------------------------
extern "C" void kernel_launch(void* const* d_in, const int* in_sizes, int n_in,
                              void* d_out, int out_size, void* d_ws,
                              size_t ws_size, hipStream_t stream) {
  const float* x  = (const float*)d_in[0];
  const float* Wq = (const float*)d_in[1];
  const float* bq = (const float*)d_in[2];
  const float* Wk = (const float*)d_in[3];
  const float* bk = (const float*)d_in[4];
  const float* Wv = (const float*)d_in[5];
  const float* bv = (const float*)d_in[6];
  const float* Wo = (const float*)d_in[7];
  const float* bo = (const float*)d_in[8];

  __bf16* qb  = (__bf16*)d_ws;                    // [B][H][L][HD]
  __bf16* kb  = qb + (size_t)MTOT * D_MODEL;      // [B][H][L][HD]
  __bf16* vb  = kb + (size_t)MTOT * D_MODEL;      // [B][H][HD][L]  (V^T)
  __bf16* ctx = vb + (size_t)MTOT * D_MODEL;      // [B][L][D]

  dim3 g1(MTOT / 128, NH, 3);
  qkv_rope_kernel<<<g1, 256, 0, stream>>>(x, Wq, bq, Wk, bk, Wv, bv, qb, kb, vb);

  dim3 g2(SEQ / 128, NH, BATCH);
  attn_kernel<<<g2, 256, 0, stream>>>(qb, kb, vb, ctx);

  dim3 g3(MTOT / 128, D_MODEL / 64, 1);
  oproj_kernel<<<g3, 256, 0, stream>>>(ctx, Wo, bo, (float*)d_out);
}
